// Encoder_57492432224437
// MI455X (gfx1250) — matrix operand order repaired and checked by reading, not compiled, not run on hardware
//
#include <hip/hip_runtime.h>
#include <hip/hip_bf16.h>

// Sizes fixed by the reference
#define BB   64
#define SS   256
#define VV   2048
#define DD   512
#define EE   256
#define TDIM 122

typedef __bf16 bf16_t;
typedef __attribute__((ext_vector_type(16))) __bf16 v16bf;
typedef __attribute__((ext_vector_type(8)))  __bf16 v8bf;
typedef __attribute__((ext_vector_type(4)))  __bf16 v4bf;
typedef __attribute__((ext_vector_type(8)))  float  v8f;

__device__ __forceinline__ bf16_t f2bf(float f){
  unsigned u = __float_as_uint(f);
  u += 0x7FFFu + ((u >> 16) & 1u);               // round-to-nearest-even
  unsigned short h = (unsigned short)(u >> 16);
  return __builtin_bit_cast(bf16_t, h);
}
__device__ __forceinline__ float bf2f(bf16_t b){
  unsigned short h = __builtin_bit_cast(unsigned short, b);
  return __uint_as_float(((unsigned)h) << 16);
}

#define SHUF16(lo, hi) __builtin_shufflevector((lo), (hi), 0,1,2,3,4,5,6,7,8,9,10,11,12,13,14,15)

// fp32 -> bf16 bulk convert (n multiple of 4)
__global__ void cvt_f32_bf16(const float* __restrict__ src, bf16_t* __restrict__ dst, long n)
{
  long i = ((long)blockIdx.x * blockDim.x + threadIdx.x) * 4;
  if (i + 3 < n){
    float4 v = *(const float4*)(src + i);
    v4bf o;
    o[0] = f2bf(v.x); o[1] = f2bf(v.y); o[2] = f2bf(v.z); o[3] = f2bf(v.w);
    *(v4bf*)(dst + i) = o;
  }
}

// w (O,I,3) fp32 -> wt (3,O,I) bf16
__global__ void pack_conv_w(const float* __restrict__ w, bf16_t* __restrict__ wt)
{
  const int D = DD;
  long idx = (long)blockIdx.x * blockDim.x + threadIdx.x;
  long total = 3L * D * D;
  if (idx >= total) return;
  int tap = (int)(idx / (D * D));
  int rem = (int)(idx % (D * D));
  int o = rem / D, i = rem % D;
  wt[idx] = f2bf(w[((long)o * D + i) * 3 + tap]);
}

// ---------------------------------------------------------------------------
// C = act(scale * A(MxK) * B(NxK)^T + bias), bf16 in, bf16 or fp32 out.
// block = 256 threads = 8 waves; wave tile = 32(M) x 64(N) (8 WMMA / k-step)
// grid = (M/256, N/64, nbatch); strides in elements.
// ---------------------------------------------------------------------------
__global__ void wmma_gemm_bf(const bf16_t* __restrict__ A, const bf16_t* __restrict__ Bm,
                             void* __restrict__ Cout, const float* __restrict__ bias,
                             int M, int N, int K,
                             long sAb, long sBb, long sCb,
                             float scale, int relu, int out_bf)
{
  int lane = threadIdx.x & 31;
  int wave = threadIdx.x >> 5;
  const bf16_t* Ab = A  + (long)blockIdx.z * sAb;
  const bf16_t* Bb = Bm + (long)blockIdx.z * sBb;
  int m0 = (blockIdx.x * 8 + wave) * 32;
  int n0 = blockIdx.y * 64;
  int kbA = (lane < 16) ? 0 : 16;   // contiguous k = 16h + s (matches B)
  int nl  = lane & 15;
  int kbB = (lane < 16) ? 0 : 16;   // contiguous k = 16h + s (matches A)
  const bf16_t* arow0 = Ab + (long)(m0 + (lane & 15)) * K;
  const bf16_t* arow1 = arow0 + (long)16 * K;
  v8f acc0[4] = {};
  v8f acc1[4] = {};
  for (int k0 = 0; k0 < K; k0 += 32){
    v8bf a0l = *(const v8bf*)(arow0 + k0 + kbA);
    v8bf a0h = *(const v8bf*)(arow0 + k0 + 8 + kbA);
    v8bf a1l = *(const v8bf*)(arow1 + k0 + kbA);
    v8bf a1h = *(const v8bf*)(arow1 + k0 + 8 + kbA);
    v16bf af0 = SHUF16(a0l, a0h);
    v16bf af1 = SHUF16(a1l, a1h);
#pragma unroll
    for (int nn = 0; nn < 4; ++nn){
      v16bf bfv = *(const v16bf*)(Bb + (long)(n0 + nn*16 + nl) * K + k0 + kbB);
      acc0[nn] = __builtin_amdgcn_wmma_f32_16x16x32_bf16(
          false, af0, false, bfv, (short)0, acc0[nn], false, false);
      acc1[nn] = __builtin_amdgcn_wmma_f32_16x16x32_bf16(
          false, af1, false, bfv, (short)0, acc1[nn], false, false);
    }
  }
  int rbase = (lane < 16) ? 0 : 8;  // C/D layout: vgpr r -> row r or r+8
  bf16_t* Cb16 = (bf16_t*)Cout + (long)blockIdx.z * sCb;
  float*  Cb32 = (float*)Cout  + (long)blockIdx.z * sCb;
#pragma unroll
  for (int half = 0; half < 2; ++half){
#pragma unroll
    for (int nn = 0; nn < 4; ++nn){
      int col = n0 + nn*16 + nl;
      float bv = bias ? bias[col] : 0.0f;
#pragma unroll
      for (int r = 0; r < 8; ++r){
        float v = (half ? acc1[nn][r] : acc0[nn][r]) * scale + bv;
        if (relu) v = fmaxf(v, 0.0f);
        long o = (long)(m0 + half*16 + rbase + r) * N + col;
        if (out_bf) Cb16[o] = f2bf(v);
        else        Cb32[o] = v;
      }
    }
  }
}

// ---------------------------------------------------------------------------
// Conv1d(k=3, pad=1) as 3 shifted-row WMMA GEMM passes; bf16 in/out, relu.
// X: (B*S, D) bf16.  Wt: (3, D, D) bf16 tap-major. grid = (B*S/256, D/64).
// ---------------------------------------------------------------------------
__global__ void wmma_conv3(const bf16_t* __restrict__ X, const bf16_t* __restrict__ Wt,
                           const float* __restrict__ bias, bf16_t* __restrict__ Y)
{
  const int D = DD;
  int lane = threadIdx.x & 31;
  int wave = threadIdx.x >> 5;
  int m0 = (blockIdx.x * 8 + wave) * 32;
  int n0 = blockIdx.y * 64;
  int rowg = m0 + (lane & 15);
  int bidx = rowg >> 8;            // S = 256 (rows m0..m0+31 same batch)
  int t0   = rowg & 255;
  int t1   = t0 + 16;
  int kbA = (lane < 16) ? 0 : 16;
  int nl  = lane & 15;
  int kbB = (lane < 16) ? 0 : 16;
  const bf16_t* Xb = X + ((long)bidx << 8) * D;
  v8f acc0[4] = {};
  v8f acc1[4] = {};
#pragma unroll
  for (int tap = 0; tap < 3; ++tap){
    int ts0 = t0 + tap - 1;
    int ts1 = t1 + tap - 1;
    bool v0 = (unsigned)ts0 < 256u;
    bool v1 = (unsigned)ts1 < 256u;
    const bf16_t* arow0 = Xb + (long)ts0 * D;
    const bf16_t* arow1 = Xb + (long)ts1 * D;
    const bf16_t* wt = Wt + (long)tap * D * D;
    for (int k0 = 0; k0 < D; k0 += 32){
      v16bf af0, af1;
      if (v0){
        v8bf lo = *(const v8bf*)(arow0 + k0 + kbA);
        v8bf hi = *(const v8bf*)(arow0 + k0 + 8 + kbA);
        af0 = SHUF16(lo, hi);
      } else af0 = (v16bf){};
      if (v1){
        v8bf lo = *(const v8bf*)(arow1 + k0 + kbA);
        v8bf hi = *(const v8bf*)(arow1 + k0 + 8 + kbA);
        af1 = SHUF16(lo, hi);
      } else af1 = (v16bf){};
#pragma unroll
      for (int nn = 0; nn < 4; ++nn){
        v16bf bfv = *(const v16bf*)(wt + (long)(n0 + nn*16 + nl) * D + k0 + kbB);
        acc0[nn] = __builtin_amdgcn_wmma_f32_16x16x32_bf16(
            false, af0, false, bfv, (short)0, acc0[nn], false, false);
        acc1[nn] = __builtin_amdgcn_wmma_f32_16x16x32_bf16(
            false, af1, false, bfv, (short)0, acc1[nn], false, false);
      }
    }
  }
  int rbase = (lane < 16) ? 0 : 8;
#pragma unroll
  for (int half = 0; half < 2; ++half){
#pragma unroll
    for (int nn = 0; nn < 4; ++nn){
      int col = n0 + nn*16 + nl;
      float bv = bias[col];
#pragma unroll
      for (int r = 0; r < 8; ++r){
        float v = (half ? acc1[nn][r] : acc0[nn][r]) + bv;
        Y[(long)(m0 + half*16 + rbase + r) * D + col] = f2bf(fmaxf(v, 0.f));
      }
    }
  }
}

// in-place row softmax over 256 columns; grid = B*S rows
__global__ void softmax256(float* __restrict__ A)
{
  __shared__ float red[256];
  int tid = threadIdx.x;
  long row = blockIdx.x;
  float v = A[row * 256 + tid];
  red[tid] = v; __syncthreads();
  for (int s = 128; s > 0; s >>= 1){ if (tid < s) red[tid] = fmaxf(red[tid], red[tid+s]); __syncthreads(); }
  float mx = red[0]; __syncthreads();
  float e = __expf(v - mx);
  red[tid] = e; __syncthreads();
  for (int s = 128; s > 0; s >>= 1){ if (tid < s) red[tid] += red[tid+s]; __syncthreads(); }
  float sm = red[0];
  A[row * 256 + tid] = e / sm;
}

__device__ __forceinline__ float blk_reduce_sum(float v, float* red)
{
  int tid = threadIdx.x;
  red[tid] = v; __syncthreads();
  for (int s = 128; s > 0; s >>= 1){
    if (tid < s) red[tid] += red[tid + s];
    __syncthreads();
  }
  float r = red[0]; __syncthreads();
  return r;
}

// ---------------------------------------------------------------------------
// Batched symmetric eigensolver: one workgroup per 256x256 Laplacian held in
// 256KB of LDS (CDNA5: up to 320KB/workgroup). Householder tridiag (parallel
// rank-2 updates) + sequential TQL1 + odd-even sort + top-122 normalize.
// ---------------------------------------------------------------------------
__global__ void eig_kernel(const float* __restrict__ Aall, float* __restrict__ out)
{
  extern __shared__ float smem[];
  float* Am  = smem;                 // 256*256
  float* vv  = Am + 65536;
  float* pp  = vv + 256;
  float* dd  = pp + 256;
  float* ee  = dd + 256;
  float* red = ee + 256;
  const int n = 256;
  int tid = threadIdx.x;
  const float* a = Aall + (long)blockIdx.x * 65536;

  // row sums of attention matrix
  {
    float s = 0.f;
    const float* row = a + (long)tid * 256;
    for (int j = 0; j < 256; ++j) s += row[j];
    red[tid] = s;
  }
  __syncthreads();
  // build symmetric Laplacian from lower triangle (eigvalsh UPLO='L')
  for (int idx = tid; idx < 65536; idx += 256){
    int i = idx >> 8, j = idx & 255;
    int ii = i > j ? i : j;
    int jj = i > j ? j : i;
    float v = -a[ii * 256 + jj];
    if (ii == jj) v += red[ii] + 1e-5f;
    Am[idx] = v;
  }
  __syncthreads();

  // Householder tridiagonalization
  for (int k = 0; k < n - 2; ++k){
    int lo = k + 1;
    float part = 0.f;
    for (int j = lo + 1 + tid; j < n; j += 256){
      float x = Am[(j << 8) + k];
      part += x * x;
    }
    float sigma = blk_reduce_sum(part, red);
    float x0 = Am[(lo << 8) + k];
    bool skip = (sigma <= 1e-32f);
    float beta = 0.f, v0 = 1.f;
    if (!skip){
      float mu = sqrtf(x0 * x0 + sigma);
      v0 = (x0 <= 0.f) ? (x0 - mu) : (-sigma / (x0 + mu));
      beta = 2.f * v0 * v0 / (sigma + v0 * v0);
      for (int j = lo + tid; j < n; j += 256)
        vv[j] = (j == lo) ? 1.f : Am[(j << 8) + k] / v0;
      __syncthreads();
      for (int j = lo + tid; j < n; j += 256){
        const float* rowj = Am + (j << 8);
        float s = 0.f;
        for (int m = lo; m < n; ++m) s += rowj[m] * vv[m];
        pp[j] = beta * s;
      }
      __syncthreads();
      float kp = 0.f;
      for (int j = lo + tid; j < n; j += 256) kp += vv[j] * pp[j];
      float kdot = blk_reduce_sum(kp, red);
      for (int j = lo + tid; j < n; j += 256)
        pp[j] -= 0.5f * beta * kdot * vv[j];
      __syncthreads();
      for (int j = lo + tid; j < n; j += 256){
        float vj = vv[j], pj = pp[j];
        float* rowj = Am + (j << 8);
        for (int m = lo; m < n; ++m)
          rowj[m] -= vj * pp[m] + pj * vv[m];
      }
    }
    if (tid == 0)
      ee[k] = skip ? x0 : (x0 - beta * (x0 + sigma / v0));
    __syncthreads();
  }
  if (tid < n) dd[tid] = Am[(tid << 8) + tid];
  if (tid == 0){
    ee[n - 2] = Am[((n - 1) << 8) + (n - 2)];
    ee[n - 1] = 0.f;
  }
  __syncthreads();

  // TQL1 (implicit-shift QL, eigenvalues only) on thread 0
  if (tid == 0){
    for (int l = 0; l < n; ++l){
      int iter = 0;
      int m;
      for (;;){
        for (m = l; m < n - 1; ++m){
          float sc = fabsf(dd[m]) + fabsf(dd[m + 1]);
          if (fabsf(ee[m]) <= 1.0e-7f * sc + 1e-30f) break;
        }
        if (m == l) break;
        if (iter++ >= 64) break;
        float g = (dd[l + 1] - dd[l]) / (2.f * ee[l]);
        float r = sqrtf(g * g + 1.f);
        float sgn = (g >= 0.f) ? r : -r;
        g = dd[m] - dd[l] + ee[l] / (g + sgn);
        float s = 1.f, c = 1.f, p = 0.f;
        bool under = false;
        for (int i = m - 1; i >= l; --i){
          float f  = s * ee[i];
          float b2 = c * ee[i];
          r = sqrtf(f * f + g * g);
          ee[i + 1] = r;
          if (r == 0.f){ dd[i + 1] -= p; ee[m] = 0.f; under = true; break; }
          s = f / r; c = g / r;
          g = dd[i + 1] - p;
          r = (dd[i] - g) * s + 2.f * c * b2;
          p = s * r;
          dd[i + 1] = g + p;
          g = c * r - b2;
        }
        if (!under){ dd[l] -= p; ee[l] = g; ee[m] = 0.f; }
      }
    }
  }
  __syncthreads();

  // ascending sort (odd-even transposition)
  for (int ph = 0; ph < n; ++ph){
    int i = 2 * tid + (ph & 1);
    if (i + 1 < n){
      float a0 = dd[i], a1 = dd[i + 1];
      if (a0 > a1){ dd[i] = a1; dd[i + 1] = a0; }
    }
    __syncthreads();
  }

  // normalize first TDIM eigenvalues, write out
  float np = 0.f;
  if (tid < TDIM){ float v = dd[tid]; np = v * v; }
  float nrm = sqrtf(blk_reduce_sum(np, red));
  nrm = fmaxf(nrm, 1e-12f);
  if (tid < TDIM) out[(long)blockIdx.x * TDIM + tid] = dd[tid] / nrm;
}

// pooled[b,d] = mean_s X[b,s,d] (X bf16);  grid (B, D/256), block 256
__global__ void pool_mean(const bf16_t* __restrict__ X, float* __restrict__ P)
{
  int b = blockIdx.x;
  int d = blockIdx.y * 256 + threadIdx.x;
  float s = 0.f;
  for (int t = 0; t < SS; ++t) s += bf2f(X[((long)b * SS + t) * DD + d]);
  P[(long)b * DD + d] = s * (1.0f / SS);
}

// H[b,o] = relu(P[b,:].W[o,:] + b1[o]); grid (B, D/256), block 256
__global__ void mlp_relu(const float* __restrict__ P, const float* __restrict__ W,
                         const float* __restrict__ b1, float* __restrict__ H)
{
  int b = blockIdx.x;
  int o = blockIdx.y * 256 + threadIdx.x;
  float s = b1[o];
  const float* pr = P + (long)b * DD;
  const float* wr = W + (long)o * DD;
  for (int d = 0; d < DD; ++d) s += pr[d] * wr[d];
  H[(long)b * DD + o] = fmaxf(s, 0.f);
}

// afmat[b,o] = softplus(H[b,:].W2[o,:] + b2[o]) + 1e-3; grid B, block 4
__global__ void afhead(const float* __restrict__ H, const float* __restrict__ W2,
                       const float* __restrict__ b2, float* __restrict__ out)
{
  int b = blockIdx.x;
  int o = threadIdx.x;
  float s = b2[o];
  const float* hr = H  + (long)b * DD;
  const float* wr = W2 + (long)o * DD;
  for (int d = 0; d < DD; ++d) s += hr[d] * wr[d];
  float sp = (s > 20.f) ? s : log1pf(__expf(s));
  out[(long)BB * TDIM + (long)b * 4 + o] = sp + 0.001f;
}

extern "C" void kernel_launch(void* const* d_in, const int* in_sizes, int n_in,
                              void* d_out, int out_size, void* d_ws, size_t ws_size,
                              hipStream_t stream)
{
  (void)in_sizes; (void)n_in; (void)out_size; (void)ws_size;
  const float* ysamp = (const float*)d_in[0];
  const float* emat  = (const float*)d_in[1];
  const float* c1w   = (const float*)d_in[2];
  const float* c1b   = (const float*)d_in[3];
  const float* c2w   = (const float*)d_in[4];
  const float* c2b   = (const float*)d_in[5];
  const float* qw    = (const float*)d_in[6];
  const float* qbv   = (const float*)d_in[7];
  const float* kw    = (const float*)d_in[8];
  const float* kbv   = (const float*)d_in[9];
  const float* a1w   = (const float*)d_in[10];
  const float* a1b   = (const float*)d_in[11];
  const float* a2w   = (const float*)d_in[12];
  const float* a2b   = (const float*)d_in[13];
  float* out = (float*)d_out;

  // workspace layout (bf16 region first, then fp32) — all 32B aligned
  const long MBS = (long)BB * SS;                    // 16384 rows
  bf16_t* ys_bf   = (bf16_t*)d_ws;                   // 33,554,432
  bf16_t* emat_bf = ys_bf + (long)MBS * VV;          //  1,048,576
  bf16_t* x0_bf   = emat_bf + (long)DD * VV;         //  8,388,608
  bf16_t* x1_bf   = x0_bf + MBS * DD;                //  8,388,608
  bf16_t* wtp_bf  = x1_bf + MBS * DD;                //    786,432
  bf16_t* qw_bf   = wtp_bf + 3L * DD * DD;           //    131,072
  bf16_t* kw_bf   = qw_bf + (long)EE * DD;           //    131,072
  float*  attn    = (float*)(kw_bf + (long)EE * DD); //  4,194,304 f32
  float*  pooled  = attn + (long)BB * SS * SS;       //     32,768
  float*  hbuf    = pooled + (long)BB * DD;          //     32,768
  bf16_t* x2_bf = x0_bf;                             // reuse
  bf16_t* qv_bf = x1_bf;                             // reuse after conv2
  bf16_t* kv_bf = x1_bf + MBS * EE;

  dim3 blk(256);

  // 0) one-time fp32 -> bf16 conversions
  cvt_f32_bf16<<<dim3((MBS * VV / 4 + 255) / 256), blk, 0, stream>>>(ysamp, ys_bf, MBS * VV);
  cvt_f32_bf16<<<dim3(((long)DD * VV / 4 + 255) / 256), blk, 0, stream>>>(emat, emat_bf, (long)DD * VV);
  cvt_f32_bf16<<<dim3(((long)EE * DD / 4 + 255) / 256), blk, 0, stream>>>(qw, qw_bf, (long)EE * DD);
  cvt_f32_bf16<<<dim3(((long)EE * DD / 4 + 255) / 256), blk, 0, stream>>>(kw, kw_bf, (long)EE * DD);

  // 1) embedding: x0 = ysamp (16384x2048) . emat^T (512x2048), bf16 out
  wmma_gemm_bf<<<dim3(16384/256, 512/64, 1), blk, 0, stream>>>(
      ys_bf, emat_bf, x0_bf, nullptr, 16384, 512, 2048, 0, 0, 0, 1.0f, 0, 1);

  // 2) conv1 (pack weights, 3-tap WMMA conv, relu, bf16 out)
  pack_conv_w<<<dim3((3*DD*DD)/256), blk, 0, stream>>>(c1w, wtp_bf);
  wmma_conv3<<<dim3(16384/256, 512/64), blk, 0, stream>>>(x0_bf, wtp_bf, c1b, x1_bf);

  // 3) conv2
  pack_conv_w<<<dim3((3*DD*DD)/256), blk, 0, stream>>>(c2w, wtp_bf);
  wmma_conv3<<<dim3(16384/256, 512/64), blk, 0, stream>>>(x1_bf, wtp_bf, c2b, x2_bf);

  // 4) q / k projections (+bias), bf16 out
  wmma_gemm_bf<<<dim3(16384/256, 256/64, 1), blk, 0, stream>>>(
      x2_bf, qw_bf, qv_bf, qbv, 16384, 256, 512, 0, 0, 0, 1.0f, 0, 1);
  wmma_gemm_bf<<<dim3(16384/256, 256/64, 1), blk, 0, stream>>>(
      x2_bf, kw_bf, kv_bf, kbv, 16384, 256, 512, 0, 0, 0, 1.0f, 0, 1);

  // 5) batched qk^T / sqrt(256), fp32 out
  wmma_gemm_bf<<<dim3(256/256, 256/64, BB), blk, 0, stream>>>(
      qv_bf, kv_bf, attn, nullptr, 256, 256, 256,
      (long)SS * EE, (long)SS * EE, (long)SS * SS, 0.0625f, 0, 0);

  // 6) softmax rows
  softmax256<<<dim3(BB * SS), blk, 0, stream>>>(attn);

  // 7) Laplacian eigenvalues -> ev output (261 KB LDS per workgroup)
  size_t eig_lds = (65536 + 5 * 256) * sizeof(float);
  eig_kernel<<<dim3(BB), blk, eig_lds, stream>>>(attn, out);

  // 8) afmat head
  pool_mean<<<dim3(BB, DD/256), blk, 0, stream>>>(x2_bf, pooled);
  mlp_relu<<<dim3(BB, DD/256), blk, 0, stream>>>(pooled, a1w, a1b, hbuf);
  afhead<<<dim3(BB), dim3(4), 0, stream>>>(hbuf, a2w, a2b, out);
}
